// Transformer_24739011625681
// MI455X (gfx1250) — compile-verified, matrix operand order checked by reading, not run on hardware
//
#include <hip/hip_runtime.h>
#include <math.h>

// ---------------- model constants ----------------
#define L_LAYERS 4
#define V_VOCAB  32000
#define D_MODEL  2304
#define F_FF     9216
#define N_HEADS  8
#define K_HEADS  4
#define HDIM     256
#define B_BATCH  2
#define T_SEQ    2048
#define BT_TOK   (B_BATCH * T_SEQ)
#define WIN      1024
#define ATTN_CAP 50.0f
#define FINAL_CAP 30.0f

typedef __attribute__((ext_vector_type(16))) _Float16 v16h;
typedef __attribute__((ext_vector_type(8)))  _Float16 v8h;
typedef __attribute__((ext_vector_type(4)))  _Float16 v4h;
typedef __attribute__((ext_vector_type(8)))  float    v8f;

union F16x16 { v16h v; v8h p[2]; _Float16 h[16]; };
union F32x8  { v8f  v; float    f[8];  };

// Per-lane K base for 16-bit A/B WMMA fragments (CDNA5 ISA):
// lanes 0-15 cover K {kb..kb+7, kb+16..kb+23} with kb=0; lanes 16-31 kb=8.
// => each lane's v16h fragment is two contiguous 8-halfword (16B) runs.
__device__ __forceinline__ int frag_kb(int lane) { return (lane & 16) ? 8 : 0; }

__device__ __forceinline__ v4h cvt4(float4 t) {
  v4h r = {(_Float16)t.x, (_Float16)t.y, (_Float16)t.z, (_Float16)t.w};
  return r;
}

// =====================================================================
// Embedding gather: x = embed[token] * sqrt(D)
// =====================================================================
__global__ __launch_bounds__(256)
void embed_kernel(const int* __restrict__ tok, const float* __restrict__ emb,
                  float* __restrict__ x) {
  long i = blockIdx.x;
  const float* e = emb + (long)tok[i] * D_MODEL;
  const float sc = sqrtf((float)D_MODEL);
  for (int d = threadIdx.x; d < D_MODEL; d += 256)
    x[i * (long)D_MODEL + d] = e[d] * sc;
}

// =====================================================================
// RMSNorm: out = in * rsqrt(mean(in^2)+eps) * (1+scale)
// If residual != nullptr: residual += norm(in) (post-norm add).
// =====================================================================
__global__ __launch_bounds__(256)
void rmsnorm_kernel(const float* __restrict__ in, const float* __restrict__ scale,
                    float* __restrict__ out, float* __restrict__ residual, int dim) {
  const long row = blockIdx.x;
  const float* x = in + row * (long)dim;
  float ss = 0.f;
  for (int d = threadIdx.x; d < dim; d += 256) { float t = x[d]; ss += t * t; }
  for (int off = 16; off > 0; off >>= 1) ss += __shfl_down(ss, off, 32);
  __shared__ float sred[8];
  int lane = threadIdx.x & 31, wid = threadIdx.x >> 5;
  if (lane == 0) sred[wid] = ss;
  __syncthreads();
  if (wid == 0) {
    float t = (lane < 8) ? sred[lane] : 0.f;
    for (int off = 4; off > 0; off >>= 1) t += __shfl_down(t, off, 32);
    if (lane == 0) sred[0] = t;
  }
  __syncthreads();
  const float inv = rsqrtf(sred[0] / (float)dim + 1e-6f);
  if (residual) {
    float* o = residual + row * (long)dim;
    for (int d = threadIdx.x; d < dim; d += 256)
      o[d] += x[d] * inv * (1.f + scale[d]);
  } else {
    float* o = out + row * (long)dim;
    for (int d = threadIdx.x; d < dim; d += 256)
      o[d] = x[d] * inv * (1.f + scale[d]);
  }
}

// =====================================================================
// Fused QK-norm + RoPE + output scaling, in place on f16 rows [rows, HDIM].
// =====================================================================
__global__ __launch_bounds__(128)
void qkprep_kernel(_Float16* __restrict__ qk, const float* __restrict__ nrm,
                   int nheads, float base, float posScale, float outScale) {
  const long row = blockIdx.x;
  const int t = (int)((row / nheads) % T_SEQ);
  _Float16* x = qk + row * (long)HDIM;
  const int j = threadIdx.x;                 // 0..127
  float x1 = (float)x[j], x2 = (float)x[j + 128];
  float ss = x1 * x1 + x2 * x2;
  for (int off = 16; off > 0; off >>= 1) ss += __shfl_down(ss, off, 32);
  __shared__ float sred[4];
  int lane = threadIdx.x & 31, wid = threadIdx.x >> 5;
  if (lane == 0) sred[wid] = ss;
  __syncthreads();
  const float tot = sred[0] + sred[1] + sred[2] + sred[3];
  const float inv = rsqrtf(tot / (float)HDIM + 1e-6f);
  x1 *= inv * (1.f + nrm[j]);
  x2 *= inv * (1.f + nrm[j + 128]);
  const float frac = 2.f * (float)j / (float)HDIM;
  const float timescale = powf(base, frac);
  const float ang = ((float)t / posScale) / timescale;
  float s, c;
  sincosf(ang, &s, &c);
  x[j]       = (_Float16)((x1 * c - x2 * s) * outScale);
  x[j + 128] = (_Float16)((x2 * c + x1 * s) * outScale);
}

// =====================================================================
// WMMA GEMM: C[M,Nc] = A[M,Kd] * B, f32 in, f16 compute, f32/f16 out.
// As row-major [m][k]; Bs TRANSPOSED [n][k] => all fragments are 2x
// ds_load_b128. Register-staged double buffering: tile k+1 is pulled from
// global into VGPRs while WMMAs consume tile k from LDS. BTRANS is a
// template parameter so the staging path is branch-free in the hot loop.
// =====================================================================
template <int BTRANS>
__global__ __launch_bounds__(128)
void gemm_f16_kernel(const float* __restrict__ A, const float* __restrict__ Bm,
                     void* __restrict__ Cv, int M, int Nc, int Kd,
                     int lda, int ldb, int ldc,
                     long aStride, long bStride, long cStride,
                     int softcap, int outF16) {
  __shared__ __align__(16) _Float16 As[64][32];   // [m][k]
  __shared__ __align__(16) _Float16 Bs[64][32];   // [n][k]  (transposed)
  const int z = blockIdx.z;
  A  += (long)z * aStride;
  Bm += (long)z * bStride;
  const int tn0 = blockIdx.x * 64;
  const int tm0 = blockIdx.y * 64;
  const int tid = threadIdx.x;
  const int lane = tid & 31, w = tid >> 5;
  const int l15 = lane & 15;
  const int kb  = frag_kb(lane);

  F32x8 acc[4];
  for (int j = 0; j < 4; j++)
    for (int r = 0; r < 8; r++) acc[j].f[r] = 0.f;

  float4 ar[4], br[4];
  auto ldA = [&](int k0) {
#pragma unroll
    for (int it = 0; it < 4; ++it) {
      const int j4 = tid + it * 128;
      const int r = j4 >> 3, kk = (j4 & 7) * 4;
      const float* src = &A[(long)(tm0 + r) * lda + (k0 + kk)];
      ar[it] = *(const float4*)src;
      __builtin_prefetch(src + 64, 0, 1);
    }
  };
  auto ldB = [&](int k0) {
#pragma unroll
    for (int it = 0; it < 4; ++it) {
      const int j4 = tid + it * 128;
      if (BTRANS == 0) {
        const int kk = j4 >> 4, nn = (j4 & 15) * 4;
        const float* src = &Bm[(long)(k0 + kk) * ldb + (tn0 + nn)];
        br[it] = *(const float4*)src;
        __builtin_prefetch(src + 64L * ldb, 0, 1);
      } else {
        const int nn = j4 >> 3, kk = (j4 & 7) * 4;
        const float* src = &Bm[(long)(tn0 + nn) * ldb + (k0 + kk)];
        br[it] = *(const float4*)src;
        __builtin_prefetch(src + 64, 0, 1);
      }
    }
  };
  auto stAB = [&]() {
#pragma unroll
    for (int it = 0; it < 4; ++it) {
      const int j4 = tid + it * 128;
      const int r = j4 >> 3, kk = (j4 & 7) * 4;
      *(v4h*)&As[r][kk] = cvt4(ar[it]);
    }
#pragma unroll
    for (int it = 0; it < 4; ++it) {
      const int j4 = tid + it * 128;
      if (BTRANS == 0) {
        const int kk = j4 >> 4, nn = (j4 & 15) * 4;
        Bs[nn + 0][kk] = (_Float16)br[it].x;
        Bs[nn + 1][kk] = (_Float16)br[it].y;
        Bs[nn + 2][kk] = (_Float16)br[it].z;
        Bs[nn + 3][kk] = (_Float16)br[it].w;
      } else {
        const int nn = j4 >> 3, kk = (j4 & 7) * 4;
        *(v4h*)&Bs[nn][kk] = cvt4(br[it]);
      }
    }
  };

  ldA(0);
  ldB(0);
  for (int k0 = 0; k0 < Kd; k0 += 32) {
    stAB();                       // waits on the in-flight loads, fills LDS
    __syncthreads();
    if (k0 + 32 < Kd) {           // issue next tile's loads; they complete
      ldA(k0 + 32);               // while the WMMAs below execute
      ldB(k0 + 32);
    }
    F16x16 af;
    const int row = w * 16 + l15;
    af.p[0] = *(const v8h*)&As[row][kb];
    af.p[1] = *(const v8h*)&As[row][kb + 16];
#pragma unroll
    for (int j = 0; j < 4; j++) {
      F16x16 bf;
      const int n = j * 16 + l15;
      bf.p[0] = *(const v8h*)&Bs[n][kb];
      bf.p[1] = *(const v8h*)&Bs[n][kb + 16];
      acc[j].v = __builtin_amdgcn_wmma_f32_16x16x32_f16(
          false, af.v, false, bf.v, (short)0, acc[j].v, false, false);
    }
    __syncthreads();
  }

  const int hf = lane >> 4;
  if (outF16) {
    _Float16* Ch = (_Float16*)Cv + (long)z * cStride;
#pragma unroll
    for (int j = 0; j < 4; j++)
#pragma unroll
      for (int r = 0; r < 8; r++) {
        long row = tm0 + w * 16 + hf * 8 + r;
        long col = tn0 + j * 16 + l15;
        if (row < M && col < Nc)
          Ch[row * (long)ldc + col] = (_Float16)acc[j].f[r];
      }
  } else {
    float* Cf = (float*)Cv + (long)z * cStride;
#pragma unroll
    for (int j = 0; j < 4; j++)
#pragma unroll
      for (int r = 0; r < 8; r++) {
        long row = tm0 + w * 16 + hf * 8 + r;
        long col = tn0 + j * 16 + l15;
        if (row < M && col < Nc) {
          float vv = acc[j].f[r];
          if (softcap) vv = FINAL_CAP * tanhf(vv / FINAL_CAP);
          Cf[row * (long)ldc + col] = vv;
        }
      }
  }
}

// =====================================================================
// Flash attention (WMMA), f16 Q/K/V: one wave = 16 query rows; 4 waves =
// 64-row block. K/V tiles (32 keys) staged once per block into LDS:
// Ks[key][h] (pure b128 copy) for QK^T B-frags, Vs[h][key] (halfword
// transpose) for PV B-frags; fragment reads are 2x ds_load_b128; Q frags
// load straight from global (2x b128, no conversion).
// grid = (T/64, N_HEADS, B).
// =====================================================================
__global__ __launch_bounds__(128)
void flash_attn_kernel(const _Float16* __restrict__ q,
                       const _Float16* __restrict__ Kp,
                       const _Float16* __restrict__ Vp,
                       float* __restrict__ out, int is_local) {
  __shared__ __align__(16) _Float16 Ks[32][HDIM];   // [key][h]
  __shared__ __align__(16) _Float16 Vs[HDIM][32];   // [h][key]
  __shared__ __align__(16) _Float16 pshare[4][16][32];
  const int qblk = blockIdx.x;
  const int n    = blockIdx.y;
  const int b    = blockIdx.z;
  const int kh   = n / (N_HEADS / K_HEADS);
  const int tid  = threadIdx.x;
  const int lane = tid & 31;
  const int w    = tid >> 5;
  const int hf   = lane >> 4;
  const int l15  = lane & 15;
  const int kb   = frag_kb(lane);
  const int qr0  = qblk * 64 + w * 16;

  // ---- Q fragments: 8 x (16x32) over HDIM=256, direct b128 loads ----
  F16x16 qf[8];
  {
    const int qrow = qr0 + l15;
    const _Float16* qp = q + ((long)(b * T_SEQ + qrow) * N_HEADS + n) * HDIM;
#pragma unroll
    for (int f = 0; f < 8; f++) {
      const _Float16* qpf = qp + f * 32 + kb;
      qf[f].p[0] = *(const v8h*)(qpf);
      qf[f].p[1] = *(const v8h*)(qpf + 16);
    }
  }

  F32x8 o[16];
  for (int j = 0; j < 16; j++)
    for (int r = 0; r < 8; r++) o[j].f[r] = 0.f;
  float rm[8], rs[8];
  for (int r = 0; r < 8; r++) { rm[r] = -1e30f; rs[r] = 0.f; }

  const int s_end = qblk * 64 + 64;
  int s_start = 0;
  if (is_local) {
    int lo = qblk * 64 - WIN + 1;
    if (lo > 0) s_start = lo & ~31;
  }
  const long kvstride = (long)K_HEADS * HDIM;
  const _Float16* kbp = Kp + (long)b * T_SEQ * kvstride + (long)kh * HDIM;
  const _Float16* vbp = Vp + (long)b * T_SEQ * kvstride + (long)kh * HDIM;

  for (int s = s_start; s < s_end; s += 32) {
    __syncthreads();   // previous iteration's reads of Ks/Vs complete
    // ---- K tile: straight f16 copy (b128 in, b128 out) ----
#pragma unroll
    for (int j8 = tid; j8 < 1024; j8 += 128) {     // 32 keys * 32 chunks of 8
      const int key = j8 >> 5;
      const int hc  = (j8 & 31) * 8;
      const _Float16* src = kbp + (long)(s + key) * kvstride + hc;
      *(v8h*)&Ks[key][hc] = *(const v8h*)src;
      __builtin_prefetch(src + 32 * kvstride, 0, 1);
    }
    // ---- V tile: halfword transpose into Vs[h][key] ----
#pragma unroll
    for (int j2 = tid; j2 < 2048; j2 += 128) {     // 32 keys * 64 chunks of 4
      const int key = j2 >> 6;
      const int hq  = (j2 & 63) * 4;
      const _Float16* src = vbp + (long)(s + key) * kvstride + hq;
      v4h u = *(const v4h*)src;
      Vs[hq + 0][key] = u[0];
      Vs[hq + 1][key] = u[1];
      Vs[hq + 2][key] = u[2];
      Vs[hq + 3][key] = u[3];
      __builtin_prefetch(src + 32 * kvstride, 0, 1);
    }
    __syncthreads();

    // ---- S = Q K^T for two 16-key subtiles ----
    F32x8 sa[2];
#pragma unroll
    for (int t2 = 0; t2 < 2; t2++) {
      for (int r = 0; r < 8; r++) sa[t2].f[r] = 0.f;
      const int krow = t2 * 16 + l15;
#pragma unroll
      for (int f = 0; f < 8; f++) {
        F16x16 kf;
        kf.p[0] = *(const v8h*)&Ks[krow][f * 32 + kb];
        kf.p[1] = *(const v8h*)&Ks[krow][f * 32 + kb + 16];
        sa[t2].v = __builtin_amdgcn_wmma_f32_16x16x32_f16(
            false, qf[f].v, false, kf.v, (short)0, sa[t2].v, false, false);
      }
    }
    // ---- soft-cap, mask, online softmax ----
    float alpha[8], pv[2][8];
#pragma unroll
    for (int r = 0; r < 8; r++) {
      const int qrow = qr0 + hf * 8 + r;
      float tmax = -1e30f;
#pragma unroll
      for (int t2 = 0; t2 < 2; t2++) {
        const int key = s + t2 * 16 + l15;
        float sv = ATTN_CAP * tanhf(sa[t2].f[r] * (1.f / ATTN_CAP));
        const bool valid = (key <= qrow) && (!is_local || (qrow - key) < WIN);
        pv[t2][r] = valid ? sv : -1e30f;
        tmax = fmaxf(tmax, pv[t2][r]);
      }
      for (int off = 1; off < 16; off <<= 1)
        tmax = fmaxf(tmax, __shfl_xor(tmax, off, 32));
      const float nm = fmaxf(rm[r], tmax);
      alpha[r] = __expf(rm[r] - nm);
      rm[r] = nm;
      float psum = 0.f;
#pragma unroll
      for (int t2 = 0; t2 < 2; t2++) {
        float p = (pv[t2][r] > -1e29f) ? __expf(pv[t2][r] - nm) : 0.f;
        pv[t2][r] = p;
        psum += p;
      }
      for (int off = 1; off < 16; off <<= 1)
        psum += __shfl_xor(psum, off, 32);
      rs[r] = rs[r] * alpha[r] + psum;
#pragma unroll
      for (int j = 0; j < 16; j++) o[j].f[r] *= alpha[r];
    }
    // ---- transpose P (D-layout -> A-layout) via per-wave LDS region ----
    // (intra-wave only: DS ops from one wave are in-order)
#pragma unroll
    for (int r = 0; r < 8; r++) {
      const int mrow = hf * 8 + r;
      pshare[w][mrow][l15]      = (_Float16)pv[0][r];
      pshare[w][mrow][16 + l15] = (_Float16)pv[1][r];
    }
    F16x16 pf;
    pf.p[0] = *(const v8h*)&pshare[w][l15][kb];
    pf.p[1] = *(const v8h*)&pshare[w][l15][kb + 16];
    // ---- O += P V ----
#pragma unroll
    for (int j = 0; j < 16; j++) {
      F16x16 vf;
      const int hh = j * 16 + l15;
      vf.p[0] = *(const v8h*)&Vs[hh][kb];
      vf.p[1] = *(const v8h*)&Vs[hh][kb + 16];
      o[j].v = __builtin_amdgcn_wmma_f32_16x16x32_f16(
          false, pf.v, false, vf.v, (short)0, o[j].v, false, false);
    }
  }

  // ---- normalize + store (f32 for the O-projection GEMM) ----
#pragma unroll
  for (int r = 0; r < 8; r++) {
    const float invs = 1.f / rs[r];
    const int mrow = hf * 8 + r;
#pragma unroll
    for (int j = 0; j < 16; j++)
      out[((long)(b * T_SEQ + qr0 + mrow) * N_HEADS + n) * HDIM + j * 16 + l15] =
          o[j].f[r] * invs;
  }
}

// =====================================================================
// GeGLU: gup[0..n) = gelu_tanh(gate) * up   (gate at [0,n), up at [n,2n))
// =====================================================================
__global__ __launch_bounds__(256)
void geglu_kernel(float* __restrict__ gup, long n) {
  long i = (long)blockIdx.x * 256 + threadIdx.x;
  if (i < n) {
    float g = gup[i], u = gup[i + n];
    float inner = 0.7978845608028654f * (g + 0.044715f * g * g * g);
    gup[i] = 0.5f * g * (1.f + tanhf(inner)) * u;
  }
}

// =====================================================================
// Host-side orchestration
// =====================================================================
static void launch_gemm(const float* A, const float* Bm, void* C,
                        int M, int Nc, int Kd, int lda, int ldb, int ldc,
                        long aS, long bS, long cS, int batches,
                        int bTrans, int softcap, int outF16, hipStream_t stream) {
  dim3 grid((Nc + 63) / 64, (M + 63) / 64, batches);
  if (bTrans)
    gemm_f16_kernel<1><<<grid, 128, 0, stream>>>(
        A, Bm, C, M, Nc, Kd, lda, ldb, ldc, aS, bS, cS, softcap, outF16);
  else
    gemm_f16_kernel<0><<<grid, 128, 0, stream>>>(
        A, Bm, C, M, Nc, Kd, lda, ldb, ldc, aS, bS, cS, softcap, outF16);
}

extern "C" void kernel_launch(void* const* d_in, const int* in_sizes, int n_in,
                              void* d_out, int out_size, void* d_ws, size_t ws_size,
                              hipStream_t stream) {
  (void)in_sizes; (void)n_in; (void)out_size; (void)ws_size;
  const int*   tokens    = (const int*)  d_in[0];
  const float* embed     = (const float*)d_in[1];
  const float* w_q       = (const float*)d_in[2];
  const float* w_kv      = (const float*)d_in[3];
  const float* w_o       = (const float*)d_in[4];
  const float* attn_norm = (const float*)d_in[5];
  const float* post_attn = (const float*)d_in[6];
  const float* ffw_norm  = (const float*)d_in[7];
  const float* post_ffw  = (const float*)d_in[8];
  const float* q_norm    = (const float*)d_in[9];
  const float* k_norm    = (const float*)d_in[10];
  const float* gate_w    = (const float*)d_in[11];
  const float* down_w    = (const float*)d_in[12];
  const float* final_nrm = (const float*)d_in[13];
  float* outp = (float*)d_out;

  // workspace layout
  float* x   = (float*)d_ws;                           // [BT, D] f32
  float* h   = x + (long)BT_TOK * D_MODEL;             // [BT, D] f32
  _Float16* qb = (_Float16*)(h + (long)BT_TOK * D_MODEL); // [BT, N*H] f16
  _Float16* kb = qb + (long)BT_TOK * N_HEADS * HDIM;   // [BT, K*H] f16
  _Float16* vb = kb + (long)BT_TOK * K_HEADS * HDIM;   // [BT, K*H] f16
  float* ao  = (float*)(vb + (long)BT_TOK * K_HEADS * HDIM); // [BT, N*H] f32
  float* tmp = ao + (long)BT_TOK * N_HEADS * HDIM;     // [BT, D] f32
  float* gup = tmp + (long)BT_TOK * D_MODEL;           // [2, BT, F] f32

  embed_kernel<<<BT_TOK, 256, 0, stream>>>(tokens, embed, x);

  for (int l = 0; l < L_LAYERS; l++) {
    const int is_local = (l % 2 == 0);
    const float base  = is_local ? 10000.f : 1000000.f;
    const float pscal = is_local ? 1.f : 8.f;

    // ---- attention block ----
    rmsnorm_kernel<<<BT_TOK, 256, 0, stream>>>(
        x, attn_norm + (long)l * D_MODEL, h, nullptr, D_MODEL);

    const float* wq_l  = w_q  + (long)l * N_HEADS * D_MODEL * HDIM;
    const float* wk_l  = w_kv + ((long)l * 2 + 0) * K_HEADS * D_MODEL * HDIM;
    const float* wv_l  = w_kv + ((long)l * 2 + 1) * K_HEADS * D_MODEL * HDIM;
    launch_gemm(h, wq_l, qb, BT_TOK, HDIM, D_MODEL, D_MODEL, HDIM,
                N_HEADS * HDIM, 0, (long)D_MODEL * HDIM, HDIM, N_HEADS,
                0, 0, /*outF16=*/1, stream);
    launch_gemm(h, wk_l, kb, BT_TOK, HDIM, D_MODEL, D_MODEL, HDIM,
                K_HEADS * HDIM, 0, (long)D_MODEL * HDIM, HDIM, K_HEADS,
                0, 0, /*outF16=*/1, stream);
    launch_gemm(h, wv_l, vb, BT_TOK, HDIM, D_MODEL, D_MODEL, HDIM,
                K_HEADS * HDIM, 0, (long)D_MODEL * HDIM, HDIM, K_HEADS,
                0, 0, /*outF16=*/1, stream);

    qkprep_kernel<<<BT_TOK * N_HEADS, 128, 0, stream>>>(
        qb, q_norm + (long)l * HDIM, N_HEADS, base, pscal,
        1.f / sqrtf((float)HDIM));
    qkprep_kernel<<<BT_TOK * K_HEADS, 128, 0, stream>>>(
        kb, k_norm + (long)l * HDIM, K_HEADS, base, pscal, 1.f);

    dim3 agrid(T_SEQ / 64, N_HEADS, B_BATCH);
    flash_attn_kernel<<<agrid, 128, 0, stream>>>(qb, kb, vb, ao, is_local);

    launch_gemm(ao, w_o + (long)l * N_HEADS * HDIM * D_MODEL, tmp,
                BT_TOK, D_MODEL, N_HEADS * HDIM,
                N_HEADS * HDIM, D_MODEL, D_MODEL, 0, 0, 0, 1, 0, 0, 0, stream);
    rmsnorm_kernel<<<BT_TOK, 256, 0, stream>>>(
        tmp, post_attn + (long)l * D_MODEL, nullptr, x, D_MODEL);

    // ---- GeGLU MLP block ----
    rmsnorm_kernel<<<BT_TOK, 256, 0, stream>>>(
        x, ffw_norm + (long)l * D_MODEL, h, nullptr, D_MODEL);
    launch_gemm(h, gate_w + (long)l * 2 * D_MODEL * F_FF, gup,
                BT_TOK, F_FF, D_MODEL, D_MODEL, F_FF, F_FF,
                0, (long)D_MODEL * F_FF, (long)BT_TOK * F_FF, 2, 0, 0, 0, stream);
    long nge = (long)BT_TOK * F_FF;
    geglu_kernel<<<(unsigned)((nge + 255) / 256), 256, 0, stream>>>(gup, nge);
    launch_gemm(gup, down_w + (long)l * F_FF * D_MODEL, tmp,
                BT_TOK, D_MODEL, F_FF, F_FF, D_MODEL, D_MODEL,
                0, 0, 0, 1, 0, 0, 0, stream);
    rmsnorm_kernel<<<BT_TOK, 256, 0, stream>>>(
        tmp, post_ffw + (long)l * D_MODEL, nullptr, x, D_MODEL);
  }

  // ---- final norm + tied decode with soft-cap ----
  rmsnorm_kernel<<<BT_TOK, 256, 0, stream>>>(x, final_nrm, h, nullptr, D_MODEL);
  launch_gemm(h, embed, outp, BT_TOK, V_VOCAB, D_MODEL,
              D_MODEL, D_MODEL, V_VOCAB, 0, 0, 0, 1, /*bTrans=*/1,
              /*softcap=*/1, /*outF16=*/0, stream);
}
